// AxialAttentionNet_74062416052443
// MI455X (gfx1250) — matrix operand order repaired and checked by reading, not compiled, not run on hardware
//
#include <hip/hip_runtime.h>
#include <cstdint>

typedef __attribute__((ext_vector_type(16))) _Float16 v16h;
typedef __attribute__((ext_vector_type(8)))  float    v8f;

#define BN_INV 0.9999950000374997f   // 1/sqrt(1+1e-5): eval-mode BN fold
#define G_ATT 8

// ============================================================================
// Generic WMMA GEMM:  C[n][m,col] = epi( A[m,:] . B[n][:,colmap(col)] )
//   A: weights [M,K] row-major f32 (f16-converted in LDS).  K % 32 == 0 always.
//   B: activations; dense fast path (identity colmap, aligned rows) uses
//      float4 loads; generic path (strided ::2 gather) uses hoisted colmap
//      with clamped, branch-free scalar loads.
//   LDS layouts are chosen so every WMMA fragment is read as contiguous
//   ds_load_b128 pairs directly into the operand register blocks.
//   OOB rows/cols are clamped (not zeroed): they only feed accumulator
//   lanes that are never stored, since K is always exact.
// Block tile 64x64, 4 waves (2x2) of 32x32, K-step 32, software-pipelined
// global->reg->LDS staging overlapping v_wmma_f32_16x16x32_f16.
// ============================================================================
struct GemmP {
  const float* A; const float* B; float* C;
  const float* scale; const float* bias; const float* add;
  int M, K, ncols;
  long sBn, sBk;                   // B strides: batch, K-row
  int  cm_w;  long cm_rs, cm_cs;   // col -> (col/cm_w)*cm_rs + (col%cm_w)*cm_cs
  long sCn, sCm, sCc;              // C strides: batch, M, col
  int relu;
  int dense;                       // identity colmap + 16B-aligned rows
};

#define LDA  40       // halfs per A row (32 + 8 pad)
#define LDBc 40       // halfs per B column (32 + 8 pad), column-major k-contig

__device__ inline v16h frag_a(const _Float16* As, int rowbase, int lane) {
  const int m = rowbase + (lane & 15);
  const int half = lane >> 4;
  v16h a;
#pragma unroll
  for (int v = 0; v < 8; ++v) {
    // A 16x32 (matches frag_b): VGPR v: K = half*16 + v*2
    int kb = (half << 4) + (v << 1);
    a[2 * v]     = As[m * LDA + kb];
    a[2 * v + 1] = As[m * LDA + kb + 1];
  }
  return a;
}

__device__ inline v16h frag_b(const _Float16* Bs, int colbase, int lane) {
  // B fragment: lane (n, half) needs K = 16*half + j, j=0..15 at column n.
  // Column-major LDS => 16 contiguous halfs = two ds_load_b128.
  const _Float16* src = &Bs[(colbase + (lane & 15)) * LDBc + ((lane >> 4) << 4)];
  v16h b;
#pragma unroll
  for (int j = 0; j < 16; ++j) b[j] = src[j];
  return b;
}

__device__ inline void store_tile(const GemmP& p, const v8f& acc,
                                  int m0, int n0, int nb, int lane) {
  const int half = lane >> 4, nn = lane & 15;
#pragma unroll
  for (int r = 0; r < 8; ++r) {
    // C/D 16x16 f32: VGPR r -> M = r + 8*half, N = lane%16
    int gm = m0 + (half << 3) + r;
    int gn = n0 + nn;
    int gmc = gm < p.M ? gm : (p.M - 1);
    int gnc = gn < p.ncols ? gn : (p.ncols - 1);
    float s = p.scale ? p.scale[gmc] * BN_INV : 1.0f;
    float b = p.bias ? p.bias[gmc] : 0.0f;
    float val = acc[r] * s + b;
    long off = (long)nb * p.sCn + (long)gmc * p.sCm + (long)gnc * p.sCc;
    if (p.add) val += p.add[off];
    if (p.relu) val = fmaxf(val, 0.0f);
    if (gm < p.M && gn < p.ncols) p.C[off] = val;
  }
}

__global__ __launch_bounds__(128) void gemm_wmma_kernel(GemmP p) {
  __shared__ _Float16 As[64 * LDA];
  __shared__ _Float16 Bs[64 * LDBc];
  const int tid  = threadIdx.x;
  const int lane = tid & 31;
  const int wid  = tid >> 5;
  const int wm = (wid >> 1) << 5;
  const int wn = (wid & 1) << 5;
  const int m0 = blockIdx.x << 6;
  const int c0 = blockIdx.y << 6;
  const int nb = blockIdx.z;
  const bool dense = p.dense != 0;

  // ---- hoisted per-thread staging addresses (constant across K-steps) ----
  const int kc4   = (tid & 7) << 2;   // A: K sub-offset, rows arow0+16i
  const int arow0 = tid >> 3;
  const int bcol4 = (tid & 15) << 2;  // B dense: 4 cols, k rows {2b,2b+1,2b+16,2b+17}
  const int bp0   = tid >> 4;         //   bp0 in 0..7
  const int gcol  = tid & 63;         // B generic: fixed col, 16 contiguous k
  const int gkb   = (tid >> 6) << 4;  //   k base 0 or 16

  long denseColBase = 0, genColOff = 0;
  if (dense) {
    int c4 = c0 + bcol4;
    int cmax = (p.ncols - 4) & ~3;
    if (c4 > cmax) c4 = cmax;
    denseColBase = (long)nb * p.sBn + c4;
  } else {
    int gc = c0 + gcol;
    int gcc = gc < p.ncols ? gc : (p.ncols - 1);
    genColOff = (long)nb * p.sBn
              + (long)(gcc / p.cm_w) * p.cm_rs + (long)(gcc % p.cm_w) * p.cm_cs;
  }

  float4 areg[4];
  float4 breg[4];

  auto load_tile = [&](int k0) {
#pragma unroll
    for (int i = 0; i < 4; ++i) {
      int gm = m0 + arow0 + (i << 4);
      if (gm >= p.M) gm = p.M - 1;
      areg[i] = *(const float4*)(p.A + (long)gm * p.K + k0 + kc4);
    }
    if (dense) {
      const float* b0 = p.B + denseColBase + (long)(k0 + 2 * bp0) * p.sBk;
      breg[0] = *(const float4*)b0;
      breg[1] = *(const float4*)(b0 + p.sBk);
      const float* b1 = b0 + 16 * p.sBk;
      breg[2] = *(const float4*)b1;
      breg[3] = *(const float4*)(b1 + p.sBk);
    } else {
#pragma unroll
      for (int r = 0; r < 16; ++r)
        ((float*)breg)[r] = p.B[genColOff + (long)(k0 + gkb + r) * p.sBk];
    }
  };

  auto stage_lds = [&]() {
#pragma unroll
    for (int i = 0; i < 4; ++i) {
      _Float16* dst = &As[(arow0 + (i << 4)) * LDA + kc4];
      dst[0] = (_Float16)areg[i].x; dst[1] = (_Float16)areg[i].y;
      dst[2] = (_Float16)areg[i].z; dst[3] = (_Float16)areg[i].w;
    }
    if (dense) {
      // per column j: k-pairs (2b,2b+1) and (2b+16,2b+17) -> two b32 stores
      const float* e0 = (const float*)&breg[0];  // row 2b   cols j
      const float* o0 = (const float*)&breg[1];  // row 2b+1
      const float* e1 = (const float*)&breg[2];  // row 2b+16
      const float* o1 = (const float*)&breg[3];  // row 2b+17
#pragma unroll
      for (int j = 0; j < 4; ++j) {
        _Float16* dst = &Bs[(bcol4 + j) * LDBc + 2 * bp0];
        dst[0]  = (_Float16)e0[j]; dst[1]  = (_Float16)o0[j];
        dst[16] = (_Float16)e1[j]; dst[17] = (_Float16)o1[j];
      }
    } else {
      // 16 contiguous k halfs of one column -> merges to b128 stores
      _Float16* dst = &Bs[gcol * LDBc + gkb];
#pragma unroll
      for (int r = 0; r < 16; ++r) dst[r] = (_Float16)((float*)breg)[r];
    }
  };

  v8f acc00 = {}; v8f acc01 = {}; v8f acc10 = {}; v8f acc11 = {};

  load_tile(0);
  for (int k0 = 0; k0 < p.K; k0 += 32) {
    stage_lds();
    __syncthreads();
    if (k0 + 32 < p.K) load_tile(k0 + 32);          // overlap next global loads
    if (k0 + 64 < p.K)
      __builtin_prefetch(p.A + (long)(m0 + arow0) * p.K + k0 + 64, 0, 1);
    v16h a0 = frag_a(As, wm, lane);
    v16h a1 = frag_a(As, wm + 16, lane);
    v16h b0 = frag_b(Bs, wn, lane);
    v16h b1 = frag_b(Bs, wn + 16, lane);
    acc00 = __builtin_amdgcn_wmma_f32_16x16x32_f16(false, a0, false, b0, (short)0, acc00, false, false);
    acc01 = __builtin_amdgcn_wmma_f32_16x16x32_f16(false, a0, false, b1, (short)0, acc01, false, false);
    acc10 = __builtin_amdgcn_wmma_f32_16x16x32_f16(false, a1, false, b0, (short)0, acc10, false, false);
    acc11 = __builtin_amdgcn_wmma_f32_16x16x32_f16(false, a1, false, b1, (short)0, acc11, false, false);
    __syncthreads();
  }

  store_tile(p, acc00, m0 + wm,      c0 + wn,      nb, lane);
  store_tile(p, acc01, m0 + wm,      c0 + wn + 16, nb, lane);
  store_tile(p, acc10, m0 + wm + 16, c0 + wn,      nb, lane);
  store_tile(p, acc11, m0 + wm + 16, c0 + wn + 16, nb, lane);
}

// ============================================================================
// Axial attention: one workgroup per (perp-col p, group g, batch n).
// q,k: [N, planes/2, H, W]; v: [N, planes, H, W] (already BN'd by GEMM).
// Fused: qr/kr rel terms + bn_qk affine + softmax(j) + sv/sve + bn_sv/bn_sve.
// ============================================================================
struct AttnP {
  const float *q, *k, *v;
  float* out;
  const float *q_rel, *k_rel, *v_rel;
  const float *qk_w, *qk_b, *qr_w, *qr_b, *kr_w, *kr_b;
  const float *sv_w, *sv_b, *sve_w, *sve_b;
  int planes, H, W, L, P;
  long stride_i, stride_p;
  int relu;
};

__global__ __launch_bounds__(64) void axial_attn_kernel(AttnP p) {
  __shared__ float qs[448], ks[448], vs[896], krs[56];
  __shared__ float sim[64 * 56];
  const int pp = blockIdx.x;
  const int g  = blockIdx.y;
  const int n  = blockIdx.z;
  const int t  = threadIdx.x;
  const int gp  = p.planes / G_ATT;
  const int gp2 = gp >> 1;
  const int L = p.L;
  const long HW = (long)p.H * p.W;
  const long pbase = (long)pp * p.stride_p;

  for (int e = t; e < gp2 * L; e += 64) {
    int c = e / L, i = e % L;
    long ch = (long)n * (p.planes >> 1) + g * gp2 + c;
    long o = ch * HW + (long)i * p.stride_i + pbase;
    qs[e] = p.q[o];
    ks[e] = p.k[o];
  }
  for (int e = t; e < gp * L; e += 64) {
    int c = e / L, i = e % L;
    long ch = (long)n * p.planes + g * gp + c;
    vs[e] = p.v[ch * HW + (long)i * p.stride_i + pbase];
  }
  __syncthreads();

  if (t < L) {
    float a = 0.0f;
    for (int c = 0; c < gp2; ++c) a += ks[c * L + t] * p.k_rel[c * L + t];
    krs[t] = a * (p.kr_w[g] * BN_INV) + p.kr_b[g];
  }
  __syncthreads();

  if (t < L) {
    const int i = t;
    const float s_qk = p.qk_w[g] * BN_INV, b_qk = p.qk_b[g];
    float qr = 0.0f;
    for (int c = 0; c < gp2; ++c) qr += qs[c * L + i] * p.q_rel[c * L + i];
    qr = qr * (p.qr_w[g] * BN_INV) + p.qr_b[g];

    float* row = &sim[t * 56];
    float mx = -1e30f;
    for (int j = 0; j < L; ++j) {
      float acc = 0.0f;
      for (int c = 0; c < gp2; ++c) acc += qs[c * L + i] * ks[c * L + j];
      float val = acc * s_qk + b_qk + qr + krs[j];
      row[j] = val;
      mx = fmaxf(mx, val);
    }
    float denom = 0.0f;
    for (int j = 0; j < L; ++j) { float e = __expf(row[j] - mx); row[j] = e; denom += e; }
    float inv = 1.0f / denom;

    for (int c = 0; c < gp; ++c) {
      float sv = 0.0f, sve = 0.0f;
      for (int j = 0; j < L; ++j) {
        float sj = row[j];
        sv  += sj * vs[c * L + j];
        sve += sj * p.v_rel[c * L + j];
      }
      sv *= inv; sve *= inv;
      int ch = g * gp + c;
      float o = sv  * (p.sv_w[ch]  * BN_INV) + p.sv_b[ch]
              + sve * (p.sve_w[ch] * BN_INV) + p.sve_b[ch];
      if (p.relu) o = fmaxf(o, 0.0f);
      p.out[((long)n * p.planes + ch) * HW + (long)i * p.stride_i + pbase] = o;
    }
  }
}

// ============================================================================
// Stem 7x7 s2 conv + BN + ReLU, maxpool 3x3 s2 p1, avgpool 2x2 s2 + ReLU,
// global avgpool (emits transposed [C,N] for the FC GEMM).
// ============================================================================
__global__ void stem_conv_kernel(const float* __restrict__ x, const float* __restrict__ w,
                                 const float* __restrict__ bw, const float* __restrict__ bb,
                                 float* __restrict__ out) {
  const int HO = 112, WO = 112, HI = 224, WI = 224, CO = 32;
  int idx = blockIdx.x * 256 + threadIdx.x;
  int total = 16 * CO * HO * WO;
  if (idx >= total) return;
  int j = idx % WO; int t2 = idx / WO;
  int i = t2 % HO;  t2 /= HO;
  int o = t2 % CO;  int n = t2 / CO;
  float acc = 0.0f;
  for (int c = 0; c < 3; ++c) {
    const float* xp = x + ((long)(n * 3 + c)) * HI * WI;
    const float* wp = w + ((o * 3 + c) * 7) * 7;
    for (int dy = 0; dy < 7; ++dy) {
      int yy = 2 * i - 3 + dy;
      if (yy < 0 || yy >= HI) continue;
      for (int dx = 0; dx < 7; ++dx) {
        int xx = 2 * j - 3 + dx;
        if (xx < 0 || xx >= WI) continue;
        acc += xp[yy * WI + xx] * wp[dy * 7 + dx];
      }
    }
  }
  out[idx] = fmaxf(acc * bw[o] * BN_INV + bb[o], 0.0f);
}

__global__ void maxpool_kernel(const float* __restrict__ in, float* __restrict__ out) {
  const int HO = 56, WO = 56, HI = 112, WI = 112;
  int idx = blockIdx.x * 256 + threadIdx.x;
  int total = 16 * 32 * HO * WO;
  if (idx >= total) return;
  int j = idx % WO; int t2 = idx / WO;
  int i = t2 % HO;  int ch = t2 / HO;
  const float* p = in + (long)ch * HI * WI;
  float m = -3.4e38f;
  for (int dy = 0; dy < 3; ++dy) {
    int yy = 2 * i - 1 + dy;
    if (yy < 0 || yy >= HI) continue;
    for (int dx = 0; dx < 3; ++dx) {
      int xx = 2 * j - 1 + dx;
      if (xx < 0 || xx >= WI) continue;
      m = fmaxf(m, p[yy * WI + xx]);
    }
  }
  out[idx] = m;
}

__global__ void avgpool2_relu_kernel(const float* __restrict__ in, float* __restrict__ out,
                                     int NC, int H, int W) {
  int HO = H >> 1, WO = W >> 1;
  int idx = blockIdx.x * 256 + threadIdx.x;
  int total = NC * HO * WO;
  if (idx >= total) return;
  int j = idx % WO; int t2 = idx / WO;
  int i = t2 % HO;  int ch = t2 / HO;
  const float* p = in + ((long)ch * H + 2 * i) * W + 2 * j;
  float v = 0.25f * (p[0] + p[1] + p[W] + p[W + 1]);
  out[idx] = fmaxf(v, 0.0f);
}

__global__ void gap_kernel(const float* __restrict__ in, float* __restrict__ outT,
                           int N, int C, int HW) {
  int idx = blockIdx.x * 256 + threadIdx.x;
  if (idx >= N * C) return;
  int c = idx % C, n = idx / C;
  const float* p = in + ((long)n * C + c) * HW;
  float s = 0.0f;
  for (int t = 0; t < HW; ++t) s += p[t];
  outT[(long)c * N + n] = s / (float)HW;   // transposed [C,N] for FC GEMM B
}

// ============================================================================
// Host side: parameter parsing (setup_inputs dict order), buffer plan, launches
// ============================================================================
struct BNp { const float *w, *b; };
struct AttW {
  const float *wq, *wk, *wv;
  BNp q, k, v, qk, qr, kr, sv, sve;
  const float *q_rel, *k_rel, *v_rel;
};
struct BlockW {
  const float* conv_down; BNp bn1;
  AttW ah, aw;
  const float* conv_up; BNp bn2;
  const float* ds_w; BNp ds_bn;
};

static inline const float* nxt(void* const* d, int& c) { return (const float*)d[c++]; }
static inline BNp getBN(void* const* d, int& c) { BNp b; b.w = nxt(d, c); b.b = nxt(d, c); return b; }
static AttW getAtt(void* const* d, int& c) {
  AttW a;
  a.wq = nxt(d, c); a.wk = nxt(d, c); a.wv = nxt(d, c);
  a.q = getBN(d, c); a.k = getBN(d, c); a.v = getBN(d, c);
  a.qk = getBN(d, c); a.qr = getBN(d, c); a.kr = getBN(d, c);
  a.sv = getBN(d, c); a.sve = getBN(d, c);
  a.q_rel = nxt(d, c); a.k_rel = nxt(d, c); a.v_rel = nxt(d, c);
  return a;
}
static BlockW getBlock(void* const* d, int& c) {
  BlockW b;
  b.conv_down = nxt(d, c); b.bn1 = getBN(d, c);
  b.ah = getAtt(d, c); b.aw = getAtt(d, c);
  b.conv_up = nxt(d, c); b.bn2 = getBN(d, c);
  b.ds_w = nxt(d, c); b.ds_bn = getBN(d, c);
  return b;
}

static void gemm_call(hipStream_t s, int batch,
                      const float* A, const float* B, float* C,
                      const float* scale, const float* bias, const float* add,
                      int M, int K, int ncols, long sBn, long sBk,
                      int cm_w, long cm_rs, long cm_cs,
                      long sCn, long sCm, long sCc, int relu) {
  GemmP g;
  g.A = A; g.B = B; g.C = C; g.scale = scale; g.bias = bias; g.add = add;
  g.M = M; g.K = K; g.ncols = ncols; g.sBn = sBn; g.sBk = sBk;
  g.cm_w = cm_w; g.cm_rs = cm_rs; g.cm_cs = cm_cs;
  g.sCn = sCn; g.sCm = sCm; g.sCc = sCc; g.relu = relu;
  g.dense = (cm_cs == 1 && cm_rs == (long)cm_w && (sBk & 3) == 0 && ncols >= 4) ? 1 : 0;
  dim3 grid((M + 63) / 64, (ncols + 63) / 64, batch);
  gemm_wmma_kernel<<<grid, dim3(128), 0, s>>>(g);
}

static void attn_call(hipStream_t s, const AttW& a, const float* q, const float* k,
                      const float* v, float* out, int planes, int H, int W,
                      int width_axis, int relu) {
  AttnP p;
  p.q = q; p.k = k; p.v = v; p.out = out;
  p.q_rel = a.q_rel; p.k_rel = a.k_rel; p.v_rel = a.v_rel;
  p.qk_w = a.qk.w; p.qk_b = a.qk.b; p.qr_w = a.qr.w; p.qr_b = a.qr.b;
  p.kr_w = a.kr.w; p.kr_b = a.kr.b;
  p.sv_w = a.sv.w; p.sv_b = a.sv.b; p.sve_w = a.sve.w; p.sve_b = a.sve.b;
  p.planes = planes; p.H = H; p.W = W;
  if (!width_axis) { p.L = H; p.P = W; p.stride_i = W; p.stride_p = 1; }
  else             { p.L = W; p.P = H; p.stride_i = 1; p.stride_p = W; }
  p.relu = relu;
  dim3 grid(p.P, G_ATT, 16);
  axial_attn_kernel<<<grid, dim3(64), 0, s>>>(p);
}

static void proj3(hipStream_t s, const AttW& a, const float* src,
                  float* qb, float* kb, float* vb, int width, long HW, int W) {
  const int N = 16;
  gemm_call(s, N, a.wq, src, qb, a.q.w, a.q.b, nullptr,
            width / 2, width, (int)HW, (long)width * HW, HW,
            W, W, 1, (long)(width / 2) * HW, HW, 1, 0);
  gemm_call(s, N, a.wk, src, kb, a.k.w, a.k.b, nullptr,
            width / 2, width, (int)HW, (long)width * HW, HW,
            W, W, 1, (long)(width / 2) * HW, HW, 1, 0);
  gemm_call(s, N, a.wv, src, vb, a.v.w, a.v.b, nullptr,
            width, width, (int)HW, (long)width * HW, HW,
            W, W, 1, (long)width * HW, HW, 1, 0);
}

// Runs one axial block; returns pointer to its output inside scratch.
static float* run_block(hipStream_t s, const float* xin, const BlockW& B,
                        int Cin, int width, int outp, int Hin, int stride, char* scr) {
  const int N = 16;
  const int H = Hin, W = Hin;
  const long HW = (long)H * W;
  const long A = (long)N * width * HW;       // elements of a full-width tensor

  float* t0   = (float*)scr;                 // [0, A)
  float* qb   = t0 + A;                      // [A, 1.5A)
  float* kb   = qb + A / 2;                  // [1.5A, 2A)
  float* vb   = kb + A / 2;                  // [2A, 3A)
  float* att1 = vb + A;                      // [3A, 4A)
  float* att2 = t0;                          // reuse t0 slot
  float* pooled = att1;                      // reuse att1 slot (stride==2)
  float* ident; float* outb;
  if (stride == 1) { ident = att1 + A; outb = ident + 2 * A; }  // [4A,6A), [6A,8A)
  else             { ident = qb;       outb = att1 + A; }       // A/2 slot, [4A,4.5A)

  // conv_down + bn1 + relu
  gemm_call(s, N, B.conv_down, xin, t0, B.bn1.w, B.bn1.b, nullptr,
            width, Cin, (int)HW, (long)Cin * HW, HW,
            W, W, 1, (long)width * HW, HW, 1, 1);

  // axial attention along H
  proj3(s, B.ah, t0, qb, kb, vb, width, HW, W);
  attn_call(s, B.ah, qb, kb, vb, att1, width, H, W, 0, 0);

  // axial attention along W (+ relu fused if no pooling)
  proj3(s, B.aw, att1, qb, kb, vb, width, HW, W);
  attn_call(s, B.aw, qb, kb, vb, att2, width, H, W, 1, stride == 1 ? 1 : 0);

  const float* xatt = att2;
  int Ho = H / stride, Wo = W / stride;
  long HWo = (long)Ho * Wo;
  if (stride == 2) {
    int tot = N * width * Ho * Wo;
    avgpool2_relu_kernel<<<(tot + 255) / 256, 256, 0, s>>>(att2, pooled, N * width, H, W);
    xatt = pooled;
  }

  // identity = ds_bn(ds_w @ x[:, :, ::stride, ::stride])
  gemm_call(s, N, B.ds_w, xin, ident, B.ds_bn.w, B.ds_bn.b, nullptr,
            outp, Cin, (int)HWo, (long)Cin * HW, HW,
            Wo, (long)stride * W, stride, (long)outp * HWo, HWo, 1, 0);

  // out = relu(bn2(conv_up @ xatt) + identity)
  gemm_call(s, N, B.conv_up, xatt, outb, B.bn2.w, B.bn2.b, ident,
            outp, width, (int)HWo, (long)width * HWo, HWo,
            Wo, Wo, 1, (long)outp * HWo, HWo, 1, 1);
  return outb;
}

extern "C" void kernel_launch(void* const* d_in, const int* in_sizes, int n_in,
                              void* d_out, int out_size, void* d_ws, size_t ws_size,
                              hipStream_t stream) {
  (void)in_sizes; (void)n_in; (void)out_size; (void)ws_size;
  int c = 0;
  const float* x = nxt(d_in, c);
  const float* stem_w = nxt(d_in, c);
  BNp stem_bn = getBN(d_in, c);
  BlockW L1 = getBlock(d_in, c);
  BlockW L2 = getBlock(d_in, c);
  BlockW L3 = getBlock(d_in, c);
  BlockW L4 = getBlock(d_in, c);
  const float* fc_w = nxt(d_in, c);
  const float* fc_b = nxt(d_in, c);

  char* ws = (char*)d_ws;
  const size_t XSLOT = (size_t)16 * 128 * 56 * 56 * sizeof(float);  // 25.7 MB input slot
  float* bufx = (float*)ws;
  char* scr = ws + XSLOT;

  // ---- stem: 7x7 s2 conv + bn + relu, then maxpool 3x3 s2 p1 -> bufx ----
  float* stem_out = (float*)scr;
  {
    int tot = 16 * 32 * 112 * 112;
    stem_conv_kernel<<<(tot + 255) / 256, 256, 0, stream>>>(x, stem_w, stem_bn.w, stem_bn.b, stem_out);
    tot = 16 * 32 * 56 * 56;
    maxpool_kernel<<<(tot + 255) / 256, 256, 0, stream>>>(stem_out, bufx);
  }

  // ---- layer1..layer4 (block output copied back into the input slot) ----
  float* o1 = run_block(stream, bufx, L1, 32, 64, 128, 56, 1, scr);
  hipMemcpyAsync(bufx, o1, (size_t)16 * 128 * 56 * 56 * sizeof(float),
                 hipMemcpyDeviceToDevice, stream);
  float* o2 = run_block(stream, bufx, L2, 128, 128, 256, 56, 2, scr);
  hipMemcpyAsync(bufx, o2, (size_t)16 * 256 * 28 * 28 * sizeof(float),
                 hipMemcpyDeviceToDevice, stream);
  float* o3 = run_block(stream, bufx, L3, 256, 256, 512, 28, 2, scr);
  hipMemcpyAsync(bufx, o3, (size_t)16 * 512 * 14 * 14 * sizeof(float),
                 hipMemcpyDeviceToDevice, stream);
  float* o4 = run_block(stream, bufx, L4, 512, 512, 1024, 14, 2, scr);

  // ---- head: global avg pool (transposed [1024,16]) + FC via WMMA GEMM ----
  float* ht = (float*)(scr + (size_t)60 * 1024 * 1024);
  {
    int tot = 16 * 1024;
    gap_kernel<<<(tot + 255) / 256, 256, 0, stream>>>(o4, ht, 16, 1024, 49);
  }
  // out[n, o] = sum_c h[n,c] * fc_w[o,c] + fc_b[o]; C strided so d_out is [16,1000]
  gemm_call(stream, 1, fc_w, ht, (float*)d_out, nullptr, fc_b, nullptr,
            1000, 1024, 16, 0, 16,
            16, 16, 1, 0, 1, 1000, 0);
}
